// RingGivensRotationLayer_57080115363916
// MI455X (gfx1250) — matrix operand order repaired and checked by reading, not compiled, not run on hardware
//
#include <hip/hip_runtime.h>

// ---------------------------------------------------------------------------
// out[b, m] = sum_k x[b, k] * R[m, k],  x: 16384 x 2048 f32,
// R = ring-Givens product built by an O(N)-per-row streaming scan.
// GEMM: bf16 WMMA 16x16x32 (f32 accum), double-buffered LDS fed by
// GLOBAL_LOAD_ASYNC_TO_LDS_B128 (ASYNCcnt) on gfx1250.
// Wave tile 64x64 (16 WMMA : 16 ds_load_b128 per K=32 chunk).
// ---------------------------------------------------------------------------

typedef __attribute__((ext_vector_type(16))) __bf16 v16bf;
typedef __attribute__((ext_vector_type(8)))  float  v8f;
typedef __attribute__((ext_vector_type(4)))  float  v4f;

#define NDIM 2048
#define BROWS 16384

#define BM 256
#define BN 128
#define BK 64
#define LDA 72        // padded LDS row stride (bf16) -> 36 dwords, conflict-free
#define LDB 72
#define ABUF_BYTES (BM * LDA * 2)   // 36864 B per A buffer
#define BBUF_BYTES (BN * LDB * 2)   // 18432 B per B buffer

__device__ __forceinline__ unsigned short f2bf(float f) {
    union { float f; unsigned u; } v; v.f = f;
    unsigned r = v.u + 0x7FFFu + ((v.u >> 16) & 1u);   // RNE
    return (unsigned short)(r >> 16);
}

__device__ __forceinline__ void async16(unsigned ldsaddr, unsigned goff, const void* sbase) {
    // GVS mode: mem = SGPR64 base + VGPR32 offset; VDST = per-lane LDS byte addr.
    asm volatile("global_load_async_to_lds_b128 %0, %1, %2"
                 :: "v"(ldsaddr), "v"(goff), "s"(sbase) : "memory");
}

__device__ __forceinline__ void wait_async0() {
#if __has_builtin(__builtin_amdgcn_s_wait_asynccnt)
    __builtin_amdgcn_s_wait_asynccnt(0);
#else
    asm volatile("s_wait_asynccnt 0x0" ::: "memory");
#endif
}

// ---------------------------------------------------------------------------
// Kernel 0: x f32 -> bf16 (one float4 per thread; NT reads: x f32 never reused)
// ---------------------------------------------------------------------------
__global__ __launch_bounds__(256) void convert_x_kernel(
    const float* __restrict__ X, unsigned short* __restrict__ Xb)
{
    const size_t i = (size_t)blockIdx.x * 256 + threadIdx.x;   // float4 index
    const v4f f = __builtin_nontemporal_load((const v4f*)X + i);
    uint2 p;
    p.x = (unsigned)f2bf(f.x) | ((unsigned)f2bf(f.y) << 16);
    p.y = (unsigned)f2bf(f.z) | ((unsigned)f2bf(f.w) << 16);
    *((uint2*)Xb + i) = p;
}

// ---------------------------------------------------------------------------
// Kernel 1: build R (row-major bf16) — exact streaming simulation of the scan.
// ---------------------------------------------------------------------------
__global__ __launch_bounds__(256) void build_R_kernel(
    const float* __restrict__ angles, unsigned short* __restrict__ R)
{
    __shared__ float cs[NDIM];
    __shared__ float sn[NDIM];
    const int t = threadIdx.x;
    for (int i = t; i < NDIM; i += 256) {
        float s, c;
        sincosf(angles[i], &s, &c);
        cs[i] = c; sn[i] = s;
    }
    __syncthreads();

    const int r = blockIdx.x * 256 + t;
    unsigned short* __restrict__ row = R + (size_t)r * NDIM;

    float v0    = 0.0f;
    float carry = (r == 0) ? 1.0f : 0.0f;

    for (int k = 0; k < NDIM; ++k) {
        const float c = cs[k];
        const float s = sn[k];
        const float vk  = carry;
        const float vk1 = (k + 1 == r) ? 1.0f : ((k == NDIM - 1) ? v0 : 0.0f);
        const float nk  = fmaf(c, vk,  s * vk1);
        const float nk1 = fmaf(c, vk1, -s * vk);
        carry = nk1;
        row[k] = f2bf(nk);
        if (k == 0) v0 = nk;         // v[0] is revisited by the wrap step
    }
    row[0] = f2bf(carry);            // final v[0]
}

// ---------------------------------------------------------------------------
// Kernel 2 (fast): async double-buffered bf16 WMMA GEMM.
// Block 256x128, 8 waves (4x2), wave tile 64x64 = 4x4 WMMA tiles.
// ---------------------------------------------------------------------------
__global__ __launch_bounds__(256) void givens_gemm_async_kernel(
    const unsigned short* __restrict__ Xb, const unsigned short* __restrict__ R,
    float* __restrict__ Out)
{
    __shared__ alignas(16) unsigned short sA[2][BM * LDA];
    __shared__ alignas(16) unsigned short sB[2][BN * LDB];

    const int t    = threadIdx.x;
    const int lane = t & 31;
    const int wave = t >> 5;
    const int wm   = wave >> 1;        // 0..3 (64-row slab)
    const int wn   = wave & 1;         // 0..1 (64-col slab)
    const int half = lane >> 4;
    const int lm   = lane & 15;

    const int blockM = blockIdx.x * BM;
    const int blockN = blockIdx.y * BN;

    // Copy slots: A = 2048 16B chunks (8/thread), B = 1024 (4/thread).
    // Identical (row, seg) geometry: B's slots are the first 4 of A's.
    unsigned gofs[8], lofs[8];
#pragma unroll
    for (int i = 0; i < 8; ++i) {
        const int idx = t + i * 256;
        const int row = idx >> 3;          // A: 0..255, B uses 0..127
        const int seg = idx & 7;           // 8 x 16B per 64-elem row
        gofs[i] = (unsigned)(row * (NDIM * 2) + seg * 16);
        lofs[i] = (unsigned)(row * (LDA * 2) + seg * 16);
    }
    const unsigned ldsA0 = (unsigned)(size_t)(&sA[0][0]);
    const unsigned ldsB0 = (unsigned)(size_t)(&sB[0][0]);

    const char* aTile = (const char*)Xb + (size_t)blockM * (NDIM * 2);
    const char* bTile = (const char*)R  + (size_t)blockN * (NDIM * 2);

    v8f acc[4][4];
#pragma unroll
    for (int a = 0; a < 4; ++a)
#pragma unroll
        for (int b = 0; b < 4; ++b) acc[a][b] = v8f{};

    union V16 { v16bf v; uint4 q[2]; };

    constexpr int NSTAGE = NDIM / BK;      // 32

    // Prologue: stage 0 into buffer 0.
#pragma unroll
    for (int i = 0; i < 8; ++i) async16(ldsA0 + lofs[i], gofs[i], aTile);
#pragma unroll
    for (int i = 0; i < 4; ++i) async16(ldsB0 + lofs[i], gofs[i], bTile);

    for (int st = 0; st < NSTAGE; ++st) {
        wait_async0();        // this wave's stage-st copies complete
        __syncthreads();      // all waves' data visible; readers of next buf retired

        if (st + 1 < NSTAGE) {
            const unsigned buf = (unsigned)((st + 1) & 1);
            const char* aB = aTile + (size_t)(st + 1) * (BK * 2);
            const char* bB = bTile + (size_t)(st + 1) * (BK * 2);
            const unsigned aL = ldsA0 + buf * ABUF_BYTES;
            const unsigned bL = ldsB0 + buf * BBUF_BYTES;
#pragma unroll
            for (int i = 0; i < 8; ++i) async16(aL + lofs[i], gofs[i], aB);
#pragma unroll
            for (int i = 0; i < 4; ++i) async16(bL + lofs[i], gofs[i], bB);
        }

        const unsigned short* curA = sA[st & 1];
        const unsigned short* curB = sB[st & 1];

#pragma unroll
        for (int kc = 0; kc < BK; kc += 32) {
            V16 afr[4], bfr[4];
#pragma unroll
            for (int tm = 0; tm < 4; ++tm) {
                const unsigned short* base =
                    curA + (wm * 64 + tm * 16 + lm) * LDA + kc + 8 * half;
                afr[tm].q[0] = *(const uint4*)(base);        // K: 8*half .. +7
                afr[tm].q[1] = *(const uint4*)(base + 16);   // K: 16+8*half .. +7
            }
#pragma unroll
            for (int tn = 0; tn < 4; ++tn) {
                const unsigned short* base =
                    curB + (wn * 64 + tn * 16 + lm) * LDB + kc + 8 * half;
                bfr[tn].q[0] = *(const uint4*)(base);
                bfr[tn].q[1] = *(const uint4*)(base + 16);
            }
#pragma unroll
            for (int tm = 0; tm < 4; ++tm)
#pragma unroll
                for (int tn = 0; tn < 4; ++tn)
                    acc[tm][tn] = __builtin_amdgcn_wmma_f32_16x16x32_bf16(
                        false, afr[tm].v, false, bfr[tn].v,
                        (short)0, acc[tm][tn], false, false);
        }
    }

    // Store (NT: keep x_bf16 / R resident in L2). C layout: M = v + 8*half, N = lm.
#pragma unroll
    for (int tm = 0; tm < 4; ++tm)
#pragma unroll
        for (int tn = 0; tn < 4; ++tn) {
            const int gcol = blockN + wn * 64 + tn * 16 + lm;
#pragma unroll
            for (int v = 0; v < 8; ++v) {
                const int grow = blockM + wm * 64 + tm * 16 + v + 8 * half;
                __builtin_nontemporal_store(acc[tm][tn][v],
                                            Out + (size_t)grow * NDIM + gcol);
            }
        }
}

// ---------------------------------------------------------------------------
// Kernel 2 (fallback, ws too small for x_bf16): f32 staging + convert in-loop.
// Block 128x128, wave tile 64x32.
// ---------------------------------------------------------------------------
__global__ __launch_bounds__(256) void givens_gemm_f32_kernel(
    const float* __restrict__ X, const unsigned short* __restrict__ R,
    float* __restrict__ Out)
{
    __shared__ alignas(16) unsigned short sA[128 * LDA];
    __shared__ alignas(16) unsigned short sB[128 * LDB];

    const int t    = threadIdx.x;
    const int lane = t & 31;
    const int wave = t >> 5;
    const int wm   = wave >> 2;
    const int wn   = wave & 3;
    const int half = lane >> 4;
    const int lm   = lane & 15;

    const int blockM = blockIdx.x * 128;
    const int blockN = blockIdx.y * 128;

    v8f acc[4][2];
#pragma unroll
    for (int a = 0; a < 4; ++a)
#pragma unroll
        for (int b = 0; b < 2; ++b) acc[a][b] = v8f{};

    union V16 { v16bf v; uint4 q[2]; };

    for (int k0 = 0; k0 < NDIM; k0 += BK) {
        __syncthreads();
#pragma unroll
        for (int i = 0; i < 8; ++i) {
            const int idx = t + i * 256;
            const int row = idx >> 4;
            const int seg = idx & 15;
            const float4 f = *(const float4*)(X + (size_t)(blockM + row) * NDIM + k0 + seg * 4);
            uint2 p;
            p.x = (unsigned)f2bf(f.x) | ((unsigned)f2bf(f.y) << 16);
            p.y = (unsigned)f2bf(f.z) | ((unsigned)f2bf(f.w) << 16);
            *(uint2*)(sA + row * LDA + seg * 4) = p;
        }
#pragma unroll
        for (int i = 0; i < 4; ++i) {
            const int idx = t + i * 256;
            const int row = idx >> 3;
            const int seg = idx & 7;
            const uint4 q = *(const uint4*)(R + (size_t)(blockN + row) * NDIM + k0 + seg * 8);
            *(uint4*)(sB + row * LDB + seg * 8) = q;
        }
        __syncthreads();

#pragma unroll
        for (int kc = 0; kc < BK; kc += 32) {
            V16 afr[4], bfr[2];
#pragma unroll
            for (int tm = 0; tm < 4; ++tm) {
                const unsigned short* base = sA + (wm * 64 + tm * 16 + lm) * LDA + kc + 8 * half;
                afr[tm].q[0] = *(const uint4*)(base);
                afr[tm].q[1] = *(const uint4*)(base + 16);
            }
#pragma unroll
            for (int tn = 0; tn < 2; ++tn) {
                const unsigned short* base = sB + (wn * 32 + tn * 16 + lm) * LDB + kc + 8 * half;
                bfr[tn].q[0] = *(const uint4*)(base);
                bfr[tn].q[1] = *(const uint4*)(base + 16);
            }
#pragma unroll
            for (int tm = 0; tm < 4; ++tm)
#pragma unroll
                for (int tn = 0; tn < 2; ++tn)
                    acc[tm][tn] = __builtin_amdgcn_wmma_f32_16x16x32_bf16(
                        false, afr[tm].v, false, bfr[tn].v,
                        (short)0, acc[tm][tn], false, false);
        }
    }

#pragma unroll
    for (int tm = 0; tm < 4; ++tm)
#pragma unroll
        for (int tn = 0; tn < 2; ++tn) {
            const int gcol = blockN + wn * 32 + tn * 16 + lm;
#pragma unroll
            for (int v = 0; v < 8; ++v) {
                const int grow = blockM + wm * 64 + tm * 16 + v + 8 * half;
                Out[(size_t)grow * NDIM + gcol] = acc[tm][tn][v];
            }
        }
}

extern "C" void kernel_launch(void* const* d_in, const int* in_sizes, int n_in,
                              void* d_out, int out_size, void* d_ws, size_t ws_size,
                              hipStream_t stream) {
    (void)in_sizes; (void)n_in; (void)out_size;
    const float* x      = (const float*)d_in[0];
    const float* angles = (const float*)d_in[1];
    float* out = (float*)d_out;

    const size_t rBytes  = (size_t)NDIM * NDIM * 2;          // 8 MB
    const size_t xbBytes = (size_t)BROWS * NDIM * 2;         // 64 MB
    unsigned short* Rbf = (unsigned short*)d_ws;
    unsigned short* Xbf = (unsigned short*)((char*)d_ws + rBytes);

    build_R_kernel<<<NDIM / 256, 256, 0, stream>>>(angles, Rbf);

    if (ws_size >= rBytes + xbBytes) {
        const size_t nf4 = (size_t)BROWS * NDIM / 4;         // 8,388,608 float4
        convert_x_kernel<<<(unsigned)(nf4 / 256), 256, 0, stream>>>(x, Xbf);
        dim3 grid(BROWS / BM, NDIM / BN);                    // 64 x 16
        givens_gemm_async_kernel<<<grid, 256, 0, stream>>>(Xbf, Rbf, out);
    } else {
        dim3 grid(BROWS / 128, NDIM / 128);                  // 128 x 16
        givens_gemm_f32_kernel<<<grid, 256, 0, stream>>>(x, Rbf, out);
    }
}
